// SIMModel_80247168959022
// MI455X (gfx1250) — matrix operand order repaired and checked by reading, not compiled, not run on hardware
//
#include <hip/hip_runtime.h>

// ---------------- CDNA5 WMMA helpers (wave32, v_wmma_f32_16x16x32_f16) ------
typedef __attribute__((ext_vector_type(16))) _Float16 v16h;
typedef __attribute__((ext_vector_type(8)))  _Float16 v8h;
typedef __attribute__((ext_vector_type(8)))  float    v8f;

#define WMMA(a,b,c) __builtin_amdgcn_wmma_f32_16x16x32_f16(false,(a),false,(b),(short)0,(c),false,false)

// ---- A fragment (16x32 f16). contiguous k order used here (B frags match):
// lane<16: m=lane, halves 0..15 -> K kbase+0..15
// lane>=16: m=lane-16, K kbase+16..31
// Fast path: f16 row-major tile, 16B-aligned rows -> two 16B vector loads.
__device__ __forceinline__ v16h frag_a_h_fast(const _Float16* base, int ld, int kbase, int lane) {
  int m = lane & 15;
  int k0 = kbase + ((lane & 16) ? 16 : 0);
  const _Float16* row = base + m * ld + k0;
  v8h lo = *(const v8h*)(row);
  v8h hi = *(const v8h*)(row + 8);
  return __builtin_shufflevector(lo, hi, 0,1,2,3,4,5,6,7,8,9,10,11,12,13,14,15);
}

// Slow A path from f32 tile (converts; used where source must stay f32)
__device__ __forceinline__ v16h frag_a_f32(const float* base, int ld, int kbase, int lane) {
  int m = lane & 15;
  int k0 = kbase + ((lane & 16) ? 16 : 0);
  const float* row = base + m * ld;
  v16h a;
#pragma unroll
  for (int i = 0; i < 8; ++i) { a[i]     = (_Float16)row[k0 + i]; }
#pragma unroll
  for (int i = 0; i < 8; ++i) { a[8 + i] = (_Float16)row[k0 + 8 + i]; }
  return a;
}

// ---- B fragment (32x16 f16): B[k][n] = W[(nbase+n)*ld + kbase + k]
// lane<16: n=lane, K kbase+0..15 ; lane>=16: n=lane-16, K kbase+16..31
__device__ __forceinline__ v16h frag_b_h16(const _Float16* W, int ld, int nbase, int kbase, int lane) {
  const _Float16* row = W + (nbase + (lane & 15)) * ld + kbase + ((lane & 16) ? 16 : 0);
  v8h lo = *(const v8h*)(row);
  v8h hi = *(const v8h*)(row + 8);
  return __builtin_shufflevector(lo, hi, 0,1,2,3,4,5,6,7,8,9,10,11,12,13,14,15);
}

// Same but contraction only valid for K<16: lanes>=16 (K 16..31) are hard zeros.
__device__ __forceinline__ v16h frag_b_h16_k16(const _Float16* W, int ld, int nbase, int lane) {
  if (lane & 16) {
    v16h z;
#pragma unroll
    for (int i = 0; i < 16; ++i) z[i] = (_Float16)0.0f;
    return z;
  }
  const _Float16* row = W + (nbase + lane) * ld;
  v8h lo = *(const v8h*)(row);
  v8h hi = *(const v8h*)(row + 8);
  return __builtin_shufflevector(lo, hi, 0,1,2,3,4,5,6,7,8,9,10,11,12,13,14,15);
}

// Pre-packed B fragment: 32B per lane, contiguous per wave.
__device__ __forceinline__ v16h frag_b_packed(const _Float16* pk, int lane) {
  return *((const v16h*)pk + lane);
}

// ---- C/D layout: VGPR r -> row (lane<16 ? r : 8+r), col = lane&15
__device__ __forceinline__ void frag_store_f16(_Float16* Y, int ld, v8f c, int lane) {
  int n = lane & 15, rb = (lane & 16) ? 8 : 0;
#pragma unroll
  for (int r = 0; r < 8; ++r) Y[(rb + r) * ld + n] = (_Float16)c[r];
}
__device__ __forceinline__ void frag_store_bias_f32(float* Y, int ld, v8f c, const float* bias, int nbase, int lane) {
  int n = lane & 15, rb = (lane & 16) ? 8 : 0;
  float bv = bias[nbase + n];
#pragma unroll
  for (int r = 0; r < 8; ++r) Y[(rb + r) * ld + n] = c[r] + bv;
}
__device__ __forceinline__ void frag_store_bias_f16(_Float16* Y, int ld, v8f c, const float* bias, int nbase, bool relu_, int lane) {
  int n = lane & 15, rb = (lane & 16) ? 8 : 0;
  float bv = bias[nbase + n];
#pragma unroll
  for (int r = 0; r < 8; ++r) { float v = c[r] + bv; if (relu_) v = fmaxf(v, 0.0f); Y[(rb + r) * ld + n] = (_Float16)v; }
}
// transposed store: element (row rb+r, col n) -> Y[n*ld + rb+r]
__device__ __forceinline__ void frag_store_bias_f16_T(_Float16* Y, int ld, v8f c, const float* bias, int nbase, int lane) {
  int n = lane & 15, rb = (lane & 16) ? 8 : 0;
  float bv = bias[nbase + n];
#pragma unroll
  for (int r = 0; r < 8; ++r) Y[n * ld + rb + r] = (_Float16)(c[r] + bv);
}

// ---------------- constants ----------------
#define BB 512
#define LONGH 1000
#define SHORTH 50
#define TOPK 50

// ---------------- prep kernels (f32 -> f16 weight mirrors) ----------------
__global__ void k_cvt_h(const float* __restrict__ src, _Float16* __restrict__ dst, int n) {
  int i = blockIdx.x * 256 + threadIdx.x;
  if (i < n) dst[i] = (_Float16)src[i];
}
// Pack W1b/W1c blocks of attn_w1 into per-lane WMMA B-fragment layout:
// index = ((nt*2+ks)*32 + lane)*16 + e ; value = w1[n*192 + (64|128) + k]
__global__ void k_pack_w1(const float* __restrict__ w1, _Float16* __restrict__ w1b_pk, _Float16* __restrict__ w1c_pk) {
  int i = blockIdx.x * 256 + threadIdx.x;
  if (i >= 4096) return;
  int e    = i & 15;
  int lane = (i >> 4) & 31;
  int ks   = (i >> 9) & 1;
  int nt   = (i >> 11) & 3;  // (i>>9)&1 is ks, bits 10.. : careful, recompute below
  // recompute cleanly:
  int frag = i >> 9;          // 0..7 = nt*2+ks
  ks = frag & 1; nt = frag >> 1;
  int n = nt * 16 + (lane & 15);
  int k = ks * 32 + ((lane & 16) ? 16 : 0) + e;
  w1b_pk[i] = (_Float16)w1[n * 192 + 64 + k];
  w1c_pk[i] = (_Float16)w1[n * 192 + 128 + k];
}

// ---------------- kernels ----------------
__global__ void k_zero(float* p, int n) {
  int i = blockIdx.x * blockDim.x + threadIdx.x;
  if (i < n) p[i] = 0.0f;
}

// target_emb = item_emb[tid] @ tproj_w^T + tproj_b ; tA = target_emb @ W1a^T + b1
__global__ void k_target(const float* __restrict__ emb, const int* __restrict__ tid,
                         const float* __restrict__ tproj_w, const float* __restrict__ tproj_b,
                         const float* __restrict__ w1, const float* __restrict__ b1,
                         float* __restrict__ target_emb, float* __restrict__ tA) {
  int b = blockIdx.x, j = threadIdx.x;
  __shared__ float e[64], tt[64];
  e[j] = emb[(size_t)tid[b] * 64 + j];
  __syncthreads();
  float acc = tproj_b[j];
  for (int k = 0; k < 64; ++k) acc += tproj_w[j * 64 + k] * e[k];
  tt[j] = acc;
  target_emb[b * 64 + j] = acc;
  __syncthreads();
  float a2 = b1[j];
  for (int k = 0; k < 64; ++k) a2 += w1[j * 192 + k] * tt[k];
  tA[b * 64 + j] = a2;
}

// Hard-search scoring: hidden = relu(tA + E@W1b^T + (E*t)@W1c^T), score = hidden.w2
__global__ void k_scores(const float* __restrict__ emb, const int* __restrict__ lids,
                         const _Float16* __restrict__ w1b_pk, const _Float16* __restrict__ w1c_pk,
                         const float* __restrict__ w2,
                         const float* __restrict__ tgt_e, const float* __restrict__ tA,
                         float* __restrict__ scores, float* __restrict__ lsum) {
  int b = blockIdx.y;
  int lbase = blockIdx.x * 128;
  int t = threadIdx.x, lane = t & 31, wv = t >> 5;
  __shared__ __align__(16) _Float16 sEh[8][16][72];
  __shared__ __align__(16) _Float16 sEt[8][16][72];
  __shared__ float sT[64], sTA[64];
  if (t < 64) { sT[t] = tgt_e[b * 64 + t]; sTA[t] = tA[b * 64 + t]; }
  __syncthreads();
  // stage one 16x64 tile per wave: E and E*t in f16 (zero-pad l >= 1000)
  {
    int r = lane >> 1, half = lane & 1;
    int l = lbase + wv * 16 + r;
    _Float16* d0 = &sEh[wv][r][half * 32];
    _Float16* d1 = &sEt[wv][r][half * 32];
    const float* tv = &sT[half * 32];
    if (l < LONGH) {
      const float* src = emb + (size_t)lids[b * LONGH + l] * 64 + half * 32;
#pragma unroll
      for (int c = 0; c < 32; ++c) { float v = src[c]; d0[c] = (_Float16)v; d1[c] = (_Float16)(v * tv[c]); }
    } else {
#pragma unroll
      for (int c = 0; c < 32; ++c) { d0[c] = (_Float16)0.0f; d1[c] = (_Float16)0.0f; }
    }
  }
  __syncthreads();
  // byproduct: sum of long embeddings for SDIM mean
  if (t < 64) {
    float s = 0.0f;
    for (int w = 0; w < 8; ++w)
#pragma unroll
      for (int r = 0; r < 16; ++r) s += (float)sEh[w][r][t];
    atomicAdd(&lsum[b * 64 + t], s);
  }
  const _Float16* Eh = &sEh[wv][0][0];
  const _Float16* Th = &sEt[wv][0][0];
  v16h aE0 = frag_a_h_fast(Eh, 72, 0, lane);
  v16h aE1 = frag_a_h_fast(Eh, 72, 32, lane);
  v16h aT0 = frag_a_h_fast(Th, 72, 0, lane);
  v16h aT1 = frag_a_h_fast(Th, 72, 32, lane);
  float sc[8];
#pragma unroll
  for (int r = 0; r < 8; ++r) sc[r] = 0.0f;
  int col = lane & 15;
#pragma unroll
  for (int nt = 0; nt < 4; ++nt) {
    v8f acc = {};
    acc = WMMA(aE0, frag_b_packed(w1b_pk + (nt * 2 + 0) * 512, lane), acc);
    acc = WMMA(aE1, frag_b_packed(w1b_pk + (nt * 2 + 1) * 512, lane), acc);
    acc = WMMA(aT0, frag_b_packed(w1c_pk + (nt * 2 + 0) * 512, lane), acc);
    acc = WMMA(aT1, frag_b_packed(w1c_pk + (nt * 2 + 1) * 512, lane), acc);
    float ta = sTA[nt * 16 + col];
    float w2v = w2[nt * 16 + col];
#pragma unroll
    for (int r = 0; r < 8; ++r) { float h = fmaxf(acc[r] + ta, 0.0f); sc[r] += h * w2v; }
  }
#pragma unroll
  for (int m = 1; m < 16; m <<= 1)
#pragma unroll
    for (int r = 0; r < 8; ++r) sc[r] += __shfl_xor(sc[r], m, 32);
  if ((lane & 15) == 0) {
    int rbase = lbase + wv * 16 + ((lane & 16) ? 8 : 0);
#pragma unroll
    for (int r = 0; r < 8; ++r) { int l = rbase + r; if (l < LONGH) scores[b * LONGH + l] = sc[r]; }
  }
}

// softmax over L, top-50, softmax of top values, weighted embedding sum
__global__ void k_hard(const float* __restrict__ scores, const int* __restrict__ lids,
                       const float* __restrict__ emb, float* __restrict__ hard) {
  int b = blockIdx.x, t = threadIdx.x;
  __shared__ float p[LONGH];
  __shared__ float rv[256];
  __shared__ int   ri[256];
  __shared__ float selv[TOPK];
  __shared__ int   seli[TOPK];
  __shared__ int   sid[TOPK];
  __shared__ float w[TOPK];
  float mx = -1e30f;
  for (int l = t; l < LONGH; l += 256) { float v = scores[b * LONGH + l]; p[l] = v; mx = fmaxf(mx, v); }
  rv[t] = mx; __syncthreads();
  for (int s = 128; s; s >>= 1) { if (t < s) rv[t] = fmaxf(rv[t], rv[t + s]); __syncthreads(); }
  mx = rv[0]; __syncthreads();
  float sm = 0.0f;
  for (int l = t; l < LONGH; l += 256) { float e = __expf(p[l] - mx); p[l] = e; sm += e; }
  rv[t] = sm; __syncthreads();
  for (int s = 128; s; s >>= 1) { if (t < s) rv[t] += rv[t + s]; __syncthreads(); }
  float inv = 1.0f / rv[0]; __syncthreads();
  for (int l = t; l < LONGH; l += 256) p[l] *= inv;
  __syncthreads();
  for (int it = 0; it < TOPK; ++it) {
    float bv = -1e30f; int bi = LONGH;
    for (int l = t; l < LONGH; l += 256) { float v = p[l]; if (v > bv || (v == bv && l < bi)) { bv = v; bi = l; } }
    rv[t] = bv; ri[t] = bi; __syncthreads();
    for (int s = 128; s; s >>= 1) {
      if (t < s) { if (rv[t + s] > rv[t] || (rv[t + s] == rv[t] && ri[t + s] < ri[t])) { rv[t] = rv[t + s]; ri[t] = ri[t + s]; } }
      __syncthreads();
    }
    if (t == 0) { selv[it] = rv[0]; seli[it] = ri[0]; p[ri[0]] = -1e30f; }
    __syncthreads();
  }
  if (t < TOPK) { sid[t] = lids[b * LONGH + seli[t]]; rv[t] = __expf(selv[t] - selv[0]); }
  else rv[t] = 0.0f;
  __syncthreads();
  for (int s = 128; s; s >>= 1) { if (t < s) rv[t] += rv[t + s]; __syncthreads(); }
  float tot2 = rv[0]; __syncthreads();
  if (t < TOPK) w[t] = __expf(selv[t] - selv[0]) / tot2;
  __syncthreads();
  if (t < 64) {
    float a = 0.0f;
    for (int k = 0; k < TOPK; ++k) a += w[k] * emb[(size_t)sid[k] * 64 + t];
    hard[b * 64 + t] = a;
  }
}

// SDIM: hash gather mean + long mean, then 64x64 matvec
__global__ void k_sdim(const int* __restrict__ lids, const float* __restrict__ hash_emb,
                       const float* __restrict__ lsum, const float* __restrict__ sdim_w,
                       const float* __restrict__ sdim_b, float* __restrict__ soft) {
  int b = blockIdx.x, t = threadIdx.x;
  __shared__ int ids[LONGH];
  __shared__ float s[64];
  for (int i = t; i < LONGH; i += 64) ids[i] = lids[b * LONGH + i];
  __syncthreads();
  int layer = t >> 4, hd = t & 15;
  const float* hb = hash_emb + (size_t)layer * 1024 * 16 + hd;
  float acc = 0.0f;
  for (int l = 0; l < LONGH; ++l) acc += hb[(ids[l] & 1023) * 16];
  s[t] = acc * (1.0f / LONGH) + lsum[b * 64 + t] * (1.0f / LONGH);
  __syncthreads();
  float o = sdim_b[t];
  for (int k = 0; k < 64; ++k) o += sdim_w[t * 64 + k] * s[k];
  soft[b * 64 + t] = o;
}

// BST transformer: one block (4 waves) per batch row.
#define LDX 65   // f32 tiles (xs, sb)
#define LDQ 72   // f16 tiles (qh, kh, vhT) -- 16B-aligned rows
__global__ void k_bst(const float* __restrict__ emb, const int* __restrict__ sids,
                      const _Float16* __restrict__ qkvw_h, const float* __restrict__ qkv_b,
                      const _Float16* __restrict__ outw_h, const float* __restrict__ out_b,
                      const float* __restrict__ ln1_g, const float* __restrict__ ln1_b,
                      const _Float16* __restrict__ ff1w_h, const float* __restrict__ ff1_b,
                      const _Float16* __restrict__ ff2w_h, const float* __restrict__ ff2_b,
                      const float* __restrict__ ln2_g, const float* __restrict__ ln2_b,
                      float* __restrict__ bst) {
  int b = blockIdx.x;
  int t = threadIdx.x, lane = t & 31, wv = t >> 5;
  __shared__ float xs[64 * LDX];
  __shared__ float sb[64 * LDX];
  __shared__ __align__(16) _Float16 qh[64 * LDQ];
  __shared__ __align__(16) _Float16 kh[64 * LDQ];
  __shared__ __align__(16) _Float16 vh[64 * LDQ];   // stores V^T: [dim][key]
  __shared__ __align__(16) _Float16 tsc[4][16][32];
  // zero pad columns of qh (read by A-frags past head width) and all of tsc
  for (int i = t; i < 64 * 8; i += 128) { int r = i >> 3, c = 64 + (i & 7); qh[r * LDQ + c] = (_Float16)0.0f; }
  for (int i = t; i < 4 * 16 * 32; i += 128) ((_Float16*)tsc)[i] = (_Float16)0.0f;
  for (int i = t; i < 64 * 64; i += 128) {
    int r = i >> 6, d = i & 63;
    float v = 0.0f;
    if (r < SHORTH) v = emb[(size_t)sids[b * SHORTH + r] * 64 + d];
    xs[r * LDX + d] = v;
  }
  __syncthreads();
  for (int l = 0; l < 2; ++l) {
    const _Float16* Wq = qkvw_h + l * 192 * 64; const float* Bq = qkv_b + l * 192;
    const _Float16* Wo = outw_h + l * 64 * 64;  const float* Bo = out_b + l * 64;
    const float* G1 = ln1_g + l * 64;           const float* B1 = ln1_b + l * 64;
    const _Float16* W1 = ff1w_h + l * 128 * 64; const float* Bf1 = ff1_b + l * 128;
    const _Float16* W2 = ff2w_h + l * 64 * 128; const float* Bf2 = ff2_b + l * 64;
    const float* G2 = ln2_g + l * 64;           const float* B2 = ln2_b + l * 64;
    // ---- QKV projection (V stored transposed) ----
    for (int mt = 0; mt < 4; ++mt) {
      v16h a0 = frag_a_f32(xs + mt * 16 * LDX, LDX, 0, lane);
      v16h a1 = frag_a_f32(xs + mt * 16 * LDX, LDX, 32, lane);
#pragma unroll
      for (int j = 0; j < 3; ++j) {
        int nt = wv + j * 4;
        v8f acc = {};
        acc = WMMA(a0, frag_b_h16(Wq, 64, nt * 16, 0, lane), acc);
        acc = WMMA(a1, frag_b_h16(Wq, 64, nt * 16, 32, lane), acc);
        if (nt < 4)      frag_store_bias_f16(qh + nt * 16 + mt * 16 * LDQ, LDQ, acc, Bq, nt * 16, false, lane);
        else if (nt < 8) frag_store_bias_f16(kh + (nt - 4) * 16 + mt * 16 * LDQ, LDQ, acc, Bq, nt * 16, false, lane);
        else             frag_store_bias_f16_T(vh + (nt - 8) * 16 * LDQ + mt * 16, LDQ, acc, Bq, nt * 16, lane);
      }
    }
    __syncthreads();
    // ---- attention; out-proj accumulated per head into registers ----
    v8f yacc[4] = {};
    int mt = wv;
    for (int h = 0; h < 4; ++h) {
      v16h aq = frag_a_h_fast(qh + mt * 16 * LDQ + h * 16, LDQ, 0, lane);
#pragma unroll
      for (int nt = 0; nt < 4; ++nt) {
        v8f acc = {};
        acc = WMMA(aq, frag_b_h16_k16(kh + h * 16, LDQ, nt * 16, lane), acc);
        int n = lane & 15, rb = (lane & 16) ? 8 : 0;
        int c = nt * 16 + n;
        float* dp = sb + mt * 16 * LDX + nt * 16;
#pragma unroll
        for (int r = 0; r < 8; ++r) {
          float v = acc[r] * 0.25f;            // 1/sqrt(HD)
          if (c >= SHORTH) v = -1e30f;         // key-padding mask
          dp[(rb + r) * LDX + n] = v;
        }
      }
      __syncthreads();
      if (t < 64) {  // row softmax over 50 valid keys
        float* row = sb + t * LDX;
        float m = -1e30f;
        for (int c = 0; c < SHORTH; ++c) m = fmaxf(m, row[c]);
        float s = 0.0f;
        for (int c = 0; c < SHORTH; ++c) { float e = __expf(row[c] - m); row[c] = e; s += e; }
        float inv = 1.0f / s;
        for (int c = 0; c < SHORTH; ++c) row[c] *= inv;
        for (int c = SHORTH; c < 64; ++c) row[c] = 0.0f;
      }
      __syncthreads();
      { // T = P @ V_h : B from transposed V (contiguous rows)
        v8f acc = {};
#pragma unroll
        for (int ks = 0; ks < 2; ++ks) {
          v16h a = frag_a_f32(sb + mt * 16 * LDX, LDX, ks * 32, lane);
          acc = WMMA(a, frag_b_h16(vh, LDQ, h * 16, ks * 32, lane), acc);
        }
        frag_store_f16(&tsc[wv][0][0], 32, acc, lane);
      }
      __syncthreads();
      { // y += T @ Wo_h^T  (K=16, zero-padded)
        v16h aT = frag_a_h_fast(&tsc[wv][0][0], 32, 0, lane);
#pragma unroll
        for (int nt = 0; nt < 4; ++nt)
          yacc[nt] = WMMA(aT, frag_b_h16_k16(Wo + h * 16, 64, nt * 16, lane), yacc[nt]);
      }
      __syncthreads();
    }
#pragma unroll
    for (int nt = 0; nt < 4; ++nt)
      frag_store_bias_f32(sb + wv * 16 * LDX + nt * 16, LDX, yacc[nt], Bo, nt * 16, lane);
    __syncthreads();
    // ---- LN1 ----
    if (t < 64) {
      float* xr = xs + t * LDX; float* yr = sb + t * LDX;
      float m = 0.0f;
      for (int d = 0; d < 64; ++d) { float v = xr[d] + yr[d]; xr[d] = v; m += v; }
      m *= (1.0f / 64.0f);
      float var = 0.0f;
      for (int d = 0; d < 64; ++d) { float dv = xr[d] - m; var += dv * dv; }
      var *= (1.0f / 64.0f);
      float inv = rsqrtf(var + 1e-5f);
      for (int d = 0; d < 64; ++d) xr[d] = (xr[d] - m) * inv * G1[d] + B1[d];
    }
    __syncthreads();
    // ---- FF1: H = relu(xs@W1^T+b1) -> qh (cols 0..63), kh (cols 64..127) ----
    for (int m2 = 0; m2 < 4; ++m2) {
      v16h a0 = frag_a_f32(xs + m2 * 16 * LDX, LDX, 0, lane);
      v16h a1 = frag_a_f32(xs + m2 * 16 * LDX, LDX, 32, lane);
#pragma unroll
      for (int j = 0; j < 2; ++j) {
        int nt = wv + j * 4;
        v8f acc = {};
        acc = WMMA(a0, frag_b_h16(W1, 64, nt * 16, 0, lane), acc);
        acc = WMMA(a1, frag_b_h16(W1, 64, nt * 16, 32, lane), acc);
        _Float16* dst = (nt < 4) ? (qh + nt * 16) : (kh + (nt - 4) * 16);
        frag_store_bias_f16(dst + m2 * 16 * LDQ, LDQ, acc, Bf1, nt * 16, true, lane);
      }
    }
    __syncthreads();
    // ---- FF2: y = H@W2^T + b2 -> sb ----
    {
      v16h ah[4];
#pragma unroll
      for (int ks = 0; ks < 4; ++ks) {
        const _Float16* ab = (ks < 2) ? qh : kh;
        ah[ks] = frag_a_h_fast(ab + mt * 16 * LDQ, LDQ, (ks & 1) * 32, lane);
      }
#pragma unroll
      for (int nt = 0; nt < 4; ++nt) {
        v8f acc = {};
#pragma unroll
        for (int ks = 0; ks < 4; ++ks)
          acc = WMMA(ah[ks], frag_b_h16(W2, 128, nt * 16, ks * 32, lane), acc);
        frag_store_bias_f32(sb + mt * 16 * LDX + nt * 16, LDX, acc, Bf2, nt * 16, lane);
      }
    }
    __syncthreads();
    // ---- LN2 ----
    if (t < 64) {
      float* xr = xs + t * LDX; float* yr = sb + t * LDX;
      float m = 0.0f;
      for (int d = 0; d < 64; ++d) { float v = xr[d] + yr[d]; xr[d] = v; m += v; }
      m *= (1.0f / 64.0f);
      float var = 0.0f;
      for (int d = 0; d < 64; ++d) { float dv = xr[d] - m; var += dv * dv; }
      var *= (1.0f / 64.0f);
      float inv = rsqrtf(var + 1e-5f);
      for (int d = 0; d < 64; ++d) xr[d] = (xr[d] - m) * inv * G2[d] + B2[d];
    }
    __syncthreads();
  }
  if (t < 64) {
    float a = 0.0f;
    for (int r = 0; r < SHORTH; ++r) a += xs[r * LDX + t];
    bst[b * 64 + t] = a * (1.0f / SHORTH);
  }
}

// fusion MLP 256 -> 256 -> 128 -> 1
__global__ void k_fusion(const float* __restrict__ user, const float* __restrict__ hard,
                         const float* __restrict__ soft, const float* __restrict__ bst,
                         const float* __restrict__ w1, const float* __restrict__ b1,
                         const float* __restrict__ w2, const float* __restrict__ b2,
                         const float* __restrict__ w3, const float* __restrict__ b3,
                         float* __restrict__ out) {
  int b = blockIdx.x, t = threadIdx.x;
  __shared__ float comb[256], h1[256], h2[128], red[256];
  int seg = t >> 6, d = t & 63;
  const float* src = (seg == 0) ? user : (seg == 1) ? hard : (seg == 2) ? soft : bst;
  comb[t] = src[b * 64 + d];
  __syncthreads();
  float a = b1[t];
  for (int k = 0; k < 256; ++k) a += w1[t * 256 + k] * comb[k];
  h1[t] = fmaxf(a, 0.0f);
  __syncthreads();
  if (t < 128) {
    float c = b2[t];
    for (int k = 0; k < 256; ++k) c += w2[t * 256 + k] * h1[k];
    h2[t] = fmaxf(c, 0.0f);
  }
  __syncthreads();
  red[t] = (t < 128) ? h2[t] * w3[t] : 0.0f;
  __syncthreads();
  for (int s = 128; s; s >>= 1) { if (t < s) red[t] += red[t + s]; __syncthreads(); }
  if (t == 0) out[b] = red[0] + b3[0];
}

// ---------------- launcher ----------------
extern "C" void kernel_launch(void* const* d_in, const int* in_sizes, int n_in,
                              void* d_out, int out_size, void* d_ws, size_t ws_size,
                              hipStream_t stream) {
  const float* user    = (const float*)d_in[0];
  const int*   tgt     = (const int*)d_in[1];
  const int*   sids    = (const int*)d_in[2];
  const int*   lids    = (const int*)d_in[3];
  const float* emb     = (const float*)d_in[4];
  const float* tproj_w = (const float*)d_in[5];
  const float* tproj_b = (const float*)d_in[6];
  const float* attn_w1 = (const float*)d_in[7];
  const float* attn_b1 = (const float*)d_in[8];
  const float* attn_w2 = (const float*)d_in[9];
  // d_in[10] attn_b2: softmax-invariant -> unused
  const float* hash_e  = (const float*)d_in[11];
  const float* sdim_w  = (const float*)d_in[12];
  const float* sdim_b  = (const float*)d_in[13];
  const float* qkv_w   = (const float*)d_in[14];
  const float* qkv_b   = (const float*)d_in[15];
  const float* out_w   = (const float*)d_in[16];
  const float* out_b   = (const float*)d_in[17];
  const float* ln1_g   = (const float*)d_in[18];
  const float* ln1_b   = (const float*)d_in[19];
  const float* ff1_w   = (const float*)d_in[20];
  const float* ff1_b   = (const float*)d_in[21];
  const float* ff2_w   = (const float*)d_in[22];
  const float* ff2_b   = (const float*)d_in[23];
  const float* ln2_g   = (const float*)d_in[24];
  const float* ln2_b   = (const float*)d_in[25];
  const float* fw1     = (const float*)d_in[26];
  const float* fb1     = (const float*)d_in[27];
  const float* fw2     = (const float*)d_in[28];
  const float* fb2     = (const float*)d_in[29];
  const float* fw3     = (const float*)d_in[30];
  const float* fb3     = (const float*)d_in[31];
  float* out = (float*)d_out;

  float* ws = (float*)d_ws;
  float* target_emb = ws;
  float* tA         = target_emb + BB * 64;
  float* scores     = tA + BB * 64;
  float* lsum       = scores + BB * LONGH;
  float* hard       = lsum + BB * 64;
  float* soft       = hard + BB * 64;
  float* bst        = soft + BB * 64;
  _Float16* hb      = (_Float16*)(bst + BB * 64);   // 32B-aligned (float offset 708608)
  _Float16* w1b_pk  = hb;                 // 4096
  _Float16* w1c_pk  = w1b_pk + 4096;      // 4096
  _Float16* qkvw_h  = w1c_pk + 4096;      // 2*192*64
  _Float16* outw_h  = qkvw_h + 2 * 192 * 64;  // 2*64*64
  _Float16* ff1w_h  = outw_h + 2 * 64 * 64;   // 2*128*64
  _Float16* ff2w_h  = ff1w_h + 2 * 128 * 64;  // 2*64*128

  // weight prep (recomputed every call -> deterministic)
  k_pack_w1<<<dim3(16), dim3(256), 0, stream>>>(attn_w1, w1b_pk, w1c_pk);
  k_cvt_h<<<dim3((2*192*64 + 255)/256), dim3(256), 0, stream>>>(qkv_w, qkvw_h, 2*192*64);
  k_cvt_h<<<dim3((2*64*64  + 255)/256), dim3(256), 0, stream>>>(out_w, outw_h, 2*64*64);
  k_cvt_h<<<dim3((2*128*64 + 255)/256), dim3(256), 0, stream>>>(ff1_w, ff1w_h, 2*128*64);
  k_cvt_h<<<dim3((2*64*128 + 255)/256), dim3(256), 0, stream>>>(ff2_w, ff2w_h, 2*64*128);

  k_zero<<<dim3((BB * 64 + 255) / 256), dim3(256), 0, stream>>>(lsum, BB * 64);
  k_target<<<dim3(BB), dim3(64), 0, stream>>>(emb, tgt, tproj_w, tproj_b, attn_w1, attn_b1, target_emb, tA);
  k_scores<<<dim3(8, BB), dim3(256), 0, stream>>>(emb, lids, w1b_pk, w1c_pk, attn_w2, target_emb, tA, scores, lsum);
  k_hard<<<dim3(BB), dim3(256), 0, stream>>>(scores, lids, emb, hard);
  k_sdim<<<dim3(BB), dim3(64), 0, stream>>>(lids, hash_e, lsum, sdim_w, sdim_b, soft);
  k_bst<<<dim3(BB), dim3(128), 0, stream>>>(emb, sids, qkvw_h, qkv_b, outw_h, out_b,
                                            ln1_g, ln1_b, ff1w_h, ff1_b, ff2w_h, ff2_b,
                                            ln2_g, ln2_b, bst);
  k_fusion<<<dim3(BB), dim3(256), 0, stream>>>(user, hard, soft, bst, fw1, fb1, fw2, fb2, fw3, fb3, out);
}
